// Sequence_44272522887758
// MI455X (gfx1250) — compile-verified, matrix operand order checked by reading, not run on hardware
//
#include <hip/hip_runtime.h>
#include <cstdint>
#include <cstddef>

// ---------------------------------------------------------------------------
// Types for WMMA bf16: D(f32 16x16) = A(bf16 16x32) * B(bf16 32x16) + C
// ---------------------------------------------------------------------------
typedef __attribute__((ext_vector_type(16))) __bf16    v16bf;
typedef __attribute__((ext_vector_type(8)))  float     v8f;
typedef __attribute__((ext_vector_type(4)))  uint32_t  u32x4;

union ABReg {
  v16bf    bf;
  u32x4    q[2];
  uint32_t u[8];
};

__device__ __host__ __forceinline__ uint16_t f2bf(float f) {
  uint32_t u = __builtin_bit_cast(uint32_t, f);
  u += 0x7FFFu + ((u >> 16) & 1u);          // round-to-nearest-even
  return (uint16_t)(u >> 16);
}
__device__ __forceinline__ float bf2f(uint16_t s) {
  return __builtin_bit_cast(float, (uint32_t)s << 16);
}
__device__ __forceinline__ float sigm(float x) {
  return 1.0f / (1.0f + __expf(-x));
}

// A tile from LDS h-buffer stored row-major [m][k] bf16.
// 16-bit A layout: lane m (0..15) / m (16..31 => +8 k offset); VGPR r holds
// k pair { (r<4 ? 2r : 16+2(r-4)) + (lane>=16 ? 8 : 0), +1 } -> two b128 reads.
__device__ __forceinline__ v16bf load_a(const uint16_t* __restrict__ base,
                                        int ke, int kc, int lane) {
  const int m   = lane & 15;
  const int grp = (lane >> 4) & 1;
  const uint16_t* p = base + m * ke + kc * 32 + grp * 8;
  ABReg r;
  r.q[0] = *reinterpret_cast<const u32x4*>(p);        // k-words 0..3
  r.q[1] = *reinterpret_cast<const u32x4*>(p + 16);   // k-words 8..11
  return r.bf;
}

// B tile pre-swizzled in global: tile = 32 lanes x 8 dwords (lane-major).
__device__ __forceinline__ v16bf load_b(const uint32_t* __restrict__ wt,
                                        int tile, int lane) {
  const uint32_t* p = wt + (size_t)tile * 256 + lane * 8;
  ABReg r;
  r.q[0] = *reinterpret_cast<const u32x4*>(p);
  r.q[1] = *reinterpret_cast<const u32x4*>(p + 4);
  return r.bf;
}

// ---------------------------------------------------------------------------
// One LSTM layer step for the wave owning hidden-column block j = wave.
// gates(16 x 4*DH) = [x | h_prev](16 x K) * Wcat^T, K split into NKIN+NKH
// chunks of 32. Four f32 accumulators (i,f,g,o) share each A tile.
// c state stays in registers (C/D layout); h_new written bf16 to LDS.
// ---------------------------------------------------------------------------
template<int DH, int NKIN, int NKH, int NBLK,
         int HOFS_PREV, int KE_PREV, int HOFS, int KE,
         int TBASE, int BOFS, bool IS_L1>
__device__ __forceinline__ void lstm_layer(
    const uint16_t* __restrict__ hb_cur,   // current-time buffer (prev layer h)
    const uint16_t* __restrict__ hb_prv,   // prev-time buffer (this layer h)
    uint16_t* __restrict__ hb_out,         // current-time buffer (write h_new)
    const uint32_t* __restrict__ wt,
    const float* __restrict__ bias,
    const float* __restrict__ w1ih,
    const float* __restrict__ xs16,
    float* __restrict__ cr,                // 8 c-state regs for this block
    int wave, int lane)
{
  if (wave < NBLK) {
    const int j   = wave;
    const int m16 = lane & 15;
    const int grp = (lane >> 4) & 1;

    v8f acc[4];
#pragma unroll
    for (int g = 0; g < 4; ++g) {
      const float bv = bias[BOFS + g * DH + 16 * j + m16];
#pragma unroll
      for (int r = 0; r < 8; ++r) acc[g][r] = bv;
    }

    if constexpr (IS_L1) {
      // din == 1: rank-1 input contribution gates[m,n] += x[m] * w_ih[n]
#pragma unroll
      for (int g = 0; g < 4; ++g) {
        const float wv = w1ih[g * DH + 16 * j + m16];
#pragma unroll
        for (int r = 0; r < 8; ++r) acc[g][r] += xs16[grp * 8 + r] * wv;
      }
    }

    constexpr int NKT = NKIN + NKH;

    // input segment: A from previous layer's h at current time
#pragma unroll
    for (int kc = 0; kc < NKIN; ++kc) {
      const v16bf a = load_a(hb_cur + HOFS_PREV, KE_PREV, kc, lane);
#pragma unroll
      for (int g = 0; g < 4; ++g) {
        const v16bf b = load_b(wt, TBASE + (j * NKT + kc) * 4 + g, lane);
        acc[g] = __builtin_amdgcn_wmma_f32_16x16x32_bf16(
            false, a, false, b, (short)0, acc[g], false, false);
      }
    }
    // hidden segment: A from this layer's h at previous time
#pragma unroll
    for (int kc = 0; kc < NKH; ++kc) {
      const v16bf a = load_a(hb_prv + HOFS, KE, kc, lane);
#pragma unroll
      for (int g = 0; g < 4; ++g) {
        const v16bf b = load_b(wt, TBASE + (j * NKT + NKIN + kc) * 4 + g, lane);
        acc[g] = __builtin_amdgcn_wmma_f32_16x16x32_bf16(
            false, a, false, b, (short)0, acc[g], false, false);
      }
    }

    // elementwise LSTM cell; C/D layout: (m = grp*8 + r, n = m16)
#pragma unroll
    for (int r = 0; r < 8; ++r) {
      const float ig = sigm(acc[0][r]);
      const float fg = sigm(acc[1][r]);
      const float gg = tanhf(acc[2][r]);
      const float og = sigm(acc[3][r]);
      const float c  = fg * cr[r] + ig * gg;
      cr[r] = c;
      const float h  = og * tanhf(c);
      hb_out[HOFS + (grp * 8 + r) * KE + 16 * j + m16] = f2bf(h);
    }
  }
}

// ---------------------------------------------------------------------------
// Persistent kernel: one workgroup per 16-batch slice, 16 waves.
// LDS: double-buffered bf16 h for all 5 layers (2 x 8192 elems = 32KB).
// Layer element offsets/widths: L1:[0,256) L2:[4096,128) L3:[6144,64)
// L4:[7168,32) L5:[7680,32 padded).
// ---------------------------------------------------------------------------
__global__ __launch_bounds__(512, 1) void lstm_persistent(
    const float* __restrict__ inp, const uint32_t* __restrict__ wt,
    const float* __restrict__ bias, const float* __restrict__ w1ih,
    const float* __restrict__ wlin, const float* __restrict__ blin,
    float* __restrict__ out, int T)
{
  __shared__ uint16_t hb[2][8192];
  __shared__ float    xs[2][16];

  const int tid  = threadIdx.x;
  const int wave = tid >> 5;
  const int lane = tid & 31;
  const int wg   = blockIdx.x;

  for (int i = tid; i < 2 * 8192; i += 512) (&hb[0][0])[i] = 0;

  float c0[8], c1[8], c2[8], c3[8], c4[8];
#pragma unroll
  for (int r = 0; r < 8; ++r) c0[r] = c1[r] = c2[r] = c3[r] = c4[r] = 0.0f;

  float wlv[16];
  float bl = 0.0f;
  if (tid < 16) {
    bl = blin[0];
#pragma unroll
    for (int k = 0; k < 16; ++k) wlv[k] = wlin[k];
  }
  __syncthreads();

  for (int t = 0; t < T; ++t) {
    const int cur = t & 1, prv = cur ^ 1;
    if (tid < 16) xs[cur][tid] = inp[(size_t)(wg * 16 + tid) * T + t];
    __syncthreads();

    // L1: dh=256, K=256 hidden only (input handled rank-1)
    lstm_layer<256, 0, 8, 16,   0,  32,    0, 256,    0,    0, true >(
        hb[cur], hb[prv], hb[cur], wt, bias, w1ih, xs[cur], c0, wave, lane);
    __syncthreads();
    // L2: dh=128, din=256
    lstm_layer<128, 8, 4,  8,   0, 256, 4096, 128,  512, 1024, false>(
        hb[cur], hb[prv], hb[cur], wt, bias, w1ih, xs[cur], c1, wave, lane);
    __syncthreads();
    // L3: dh=64, din=128
    lstm_layer< 64, 4, 2,  4, 4096, 128, 6144,  64,  896, 1536, false>(
        hb[cur], hb[prv], hb[cur], wt, bias, w1ih, xs[cur], c2, wave, lane);
    __syncthreads();
    // L4: dh=32, din=64
    lstm_layer< 32, 2, 1,  2, 6144,  64, 7168,  32,  992, 1792, false>(
        hb[cur], hb[prv], hb[cur], wt, bias, w1ih, xs[cur], c3, wave, lane);
    __syncthreads();
    // L5: dh=16 (K padded 16->32), din=32
    lstm_layer< 16, 1, 1,  1, 7168,  32, 7680,  32, 1016, 1920, false>(
        hb[cur], hb[prv], hb[cur], wt, bias, w1ih, xs[cur], c4, wave, lane);
    __syncthreads();

    if (tid < 16) {
      float s = bl;
#pragma unroll
      for (int k = 0; k < 16; ++k)
        s += bf2f(hb[cur][7680 + tid * 32 + k]) * wlv[k];
      out[(size_t)(wg * 16 + tid) * T + t] = s;
    }
  }
}

// ---------------------------------------------------------------------------
// Preprocessing: fp32 weights -> bf16 B-tiles in the exact WMMA lane/VGPR
// swizzle. One block per 32x16 tile; thread = (lane l, vgpr r) word.
// Wcat = [W_ih | W_hh] over padded K; B[k][n] = Wcat[gate_row(n), k].
// ---------------------------------------------------------------------------
__global__ void pack_weights(const float* __restrict__ wih,
                             const float* __restrict__ whh,
                             uint32_t* __restrict__ wt,
                             int dh, int din, int nkin, int nkt, int tbase)
{
  const int tile = blockIdx.x;     // layer-local tile index
  const int tid  = threadIdx.x;    // 0..255
  const int l  = tid >> 3;         // lane 0..31
  const int r  = tid & 7;          // vgpr 0..7
  const int n  = l & 15;
  const int k0 = ((r < 4) ? 2 * r : 16 + 2 * (r - 4)) + ((l & 16) ? 8 : 0);

  const int j   = tile / (4 * nkt);
  const int rem = tile % (4 * nkt);
  const int kc  = rem >> 2;
  const int g   = rem & 3;
  const int row = g * dh + 16 * j + n;
  const int kg  = kc * 32 + k0;

  float v[2];
#pragma unroll
  for (int s = 0; s < 2; ++s) {
    const int k = kg + s;
    float x;
    if (k < nkin * 32) {
      x = (k < din) ? wih[(size_t)row * din + k] : 0.0f;
    } else {
      const int kh = k - nkin * 32;
      x = (kh < dh) ? whh[(size_t)row * dh + kh] : 0.0f;
    }
    v[s] = x;
  }
  wt[(size_t)(tbase + tile) * 256 + l * 8 + r] =
      (uint32_t)f2bf(v[0]) | ((uint32_t)f2bf(v[1]) << 16);
}

__global__ void add2(const float* __restrict__ a, const float* __restrict__ b,
                     float* __restrict__ dst, int n) {
  const int i = blockIdx.x * blockDim.x + threadIdx.x;
  if (i < n) dst[i] = a[i] + b[i];
}
__global__ void copy1(const float* __restrict__ a, float* __restrict__ dst,
                      int n) {
  const int i = blockIdx.x * blockDim.x + threadIdx.x;
  if (i < n) dst[i] = a[i];
}

// ---------------------------------------------------------------------------
// d_in order: 0 inputData; per layer l=1..5 at 1+(l-1)*4: w_ih,w_hh,b_ih,b_hh;
// 21 w_lin; 22 b_lin.  d_ws: [0,1MB) bf16 weight tiles; then bias (2048 f32);
// then w1_ih copy (1024 f32).
// ---------------------------------------------------------------------------
extern "C" void kernel_launch(void* const* d_in, const int* in_sizes, int n_in,
                              void* d_out, int out_size, void* d_ws,
                              size_t ws_size, hipStream_t stream)
{
  (void)n_in; (void)out_size; (void)ws_size;

  const float* inp  = (const float*)d_in[0];
  const float* wlin = (const float*)d_in[21];
  const float* blin = (const float*)d_in[22];
  const int B = 128;
  const int T = in_sizes[0] / B;   // 2048

  uint32_t* wt   = (uint32_t*)d_ws;                          // 1024 tiles * 1KB
  float*    bias = (float*)((char*)d_ws + 1024u * 1024u);    // 1984 used
  float*    w1ih = bias + 2048;                              // 1024 f32

  const int dh[5]    = {256, 128, 64, 32, 16};
  const int din[5]   = {  1, 256, 128, 64, 32};
  const int nkin[5]  = {  0,   8,   4,  2,  1};
  const int nkh[5]   = {  8,   4,   2,  1,  1};
  const int nblk[5]  = { 16,   8,   4,  2,  1};
  const int tbase[5] = {  0, 512, 896, 992, 1016};
  const int bofs[5]  = {  0, 1024, 1536, 1792, 1920};

  for (int l = 0; l < 5; ++l) {
    const float* wih = (const float*)d_in[1 + l * 4 + 0];
    const float* whh = (const float*)d_in[1 + l * 4 + 1];
    const float* bih = (const float*)d_in[1 + l * 4 + 2];
    const float* bhh = (const float*)d_in[1 + l * 4 + 3];
    const int ntiles = nblk[l] * 4 * (nkin[l] + nkh[l]);
    pack_weights<<<ntiles, 256, 0, stream>>>(wih, whh, wt, dh[l], din[l],
                                             nkin[l], nkin[l] + nkh[l],
                                             tbase[l]);
    const int n = 4 * dh[l];
    add2<<<(n + 255) / 256, 256, 0, stream>>>(bih, bhh, bias + bofs[l], n);
  }
  copy1<<<4, 256, 0, stream>>>((const float*)d_in[1], w1ih, 1024);

  lstm_persistent<<<B / 16, 512, 0, stream>>>(inp, wt, bias, w1ih, wlin, blin,
                                              (float*)d_out, T);
}
